// Versor_Rotor_12713103196441
// MI455X (gfx1250) — compile-verified, matrix operand order checked by reading, not run on hardware
//
#include <hip/hip_runtime.h>
#include <hip/hip_bf16.h>
#include <math.h>
#include <stdint.h>

typedef __attribute__((ext_vector_type(16))) _Float16 v16h;
typedef __attribute__((ext_vector_type(8)))  float    v8f;

// ---------------------------------------------------------------------------
// Problem constants (from reference): B=128, T=512, D=64, 32 blades.
#define NB     128
#define NT_SEQ 512
#define ND     64
#define BB     16        // batches per workgroup
#define NWG    (NB / BB) // 8 workgroups

// LDS layout (bytes). hH holds 65 rows per batch: rows 0..63 = h (f16 mirror),
// row 64 = r_t (f16) so the gate GEMM A-fragment address is uniform over all
// 65 K-tiles (gate_in = [h flat | r]).
#define HH_OFF  0                         // [BB][65][32] f16  = 66560
#define HF_OFF  66560                     // [BB][64][32] f32  = 131072
#define RS_OFF  197632                    // [2][BB][32] f32   = 4096 (TDM dbl buf)
#define SS_OFF  201728                    // sign [32][32] f32 = 4096
#define GS_OFF  205824                    // g [BB][64] f32    = 4096
#define GP_OFF  209920                    // partials [2][BB][64] / z [BB][128] = 8192
#define GB_OFF  218112                    // gate_b mirror [64] f32 = 256
#define SMEM_BYTES 218368

// ---- compile-safe feature guards ----
#if defined(__has_builtin)
#if __has_builtin(__builtin_amdgcn_tensor_load_to_lds) && \
    __has_builtin(__builtin_amdgcn_s_wait_tensorcnt)
#define HAVE_TDM 1
#endif
#if __has_builtin(__builtin_amdgcn_rcpf)
#define HAVE_RCPF 1
#endif
#endif
#ifndef HAVE_TDM
#define HAVE_TDM 0
#endif
#ifndef HAVE_RCPF
#define HAVE_RCPF 0
#endif

typedef __attribute__((ext_vector_type(4))) unsigned int tdm_u32x4;
typedef __attribute__((ext_vector_type(8))) int          tdm_i32x8;
typedef __attribute__((ext_vector_type(4))) int          tdm_i32x4;

__device__ __forceinline__ float fast_rcp(float x) {
#if HAVE_RCPF
    return __builtin_amdgcn_rcpf(x);   // v_rcp_f32 (gate/norm tolerate ~1ulp)
#else
    return 1.f / x;
#endif
}

// WMMA 16x16x32 f16 fragment index map (ISA 7.12.2, 16-bit A 16x32):
// lane l: row=(l&15); hi=l>>4 selects K halves; element e ->
// K = (e<8 ? e : e+8) + 8*hi (two contiguous 8-element runs -> 2x b128 loads)
__device__ __forceinline__ int kfrag(int e, int hi) {
    return (e < 8 ? e : e + 8) + 8 * hi;
}

__device__ __forceinline__ v8f wmma_f16(v16h a, v16h b, v8f c) {
    return __builtin_amdgcn_wmma_f32_16x16x32_f16(
        /*neg_a=*/false, a, /*neg_b=*/false, b,
        /*c_mod=*/(short)0, c, /*reuse_a=*/false, /*reuse_b=*/false);
}

union FragU { v16h v; uint4 q[2]; };

// A fragment from LDS h mirror; row = absolute row index in hH (two b128 loads)
__device__ __forceinline__ FragU loadArow(const _Float16* hH, int row, int hi) {
    const _Float16* p = hH + (size_t)row * 32;
    FragU ua;
    ua.q[0] = *(const uint4*)(p + 8 * hi);
    ua.q[1] = *(const uint4*)(p + 16 + 8 * hi);
    return ua;
}
// B fragment from fragment-ready packed weights (two b128 loads)
__device__ __forceinline__ FragU loadB(const _Float16* pack, int tileIdx, int lane) {
    const _Float16* bp = pack + ((size_t)tileIdx * 32 + lane) * 16;
    FragU ub;
    ub.q[0] = *(const uint4*)(bp);
    ub.q[1] = *(const uint4*)(bp + 8);
    return ub;
}

#if HAVE_TDM
// TDM 2D tile load: 16 rows (batches, stride rowStrideElems f32) x 32 f32
// from global r_ws into LDS byte offset lds_off.  D# per ISA ch.8.
__device__ __forceinline__ void tdm_load_r(const float* gsrc, unsigned lds_off,
                                           int rowStrideElems) {
    unsigned long long ga = (unsigned long long)(uintptr_t)gsrc;
    tdm_u32x4 g0;
    g0[0] = 1u;                                            // count=1, user D#
    g0[1] = lds_off;                                       // lds_addr (bytes)
    g0[2] = (unsigned)(ga & 0xffffffffu);                  // global_addr[31:0]
    g0[3] = (unsigned)((ga >> 32) & 0x01ffffffu) | (2u << 30); // ga[56:32] | type=2
    tdm_i32x8 g1;
    g1[0] = (int)(2u << 16);          // wg_mask=0 | data_size=2 (4 bytes)
    g1[1] = (int)(32u << 16);         // tensor_dim0 = 32
    g1[2] = (int)(16u << 16);         // tensor_dim1 = 16
    g1[3] = (int)(32u << 16);         // tile_dim0 = 32
    g1[4] = (int)(16u);               // tile_dim1 = 16 | tile_dim2 = 0
    g1[5] = rowStrideElems;           // tensor_dim0_stride[31:0]
    g1[6] = 0;
    g1[7] = 0;
    tdm_i32x4 gz = {0, 0, 0, 0};
#if defined(__clang_major__) && (__clang_major__ >= 23)
    tdm_i32x8 gz8 = {0, 0, 0, 0, 0, 0, 0, 0};
    __builtin_amdgcn_tensor_load_to_lds(g0, g1, gz, gz, gz8, 0);
#else
    __builtin_amdgcn_tensor_load_to_lds(g0, g1, gz, gz, 0);
#endif
}
#endif

// ---------------------------------------------------------------------------
// Kernel 1: r_in = normalize(emb[x]) for all (b,t) -> ws [B*T][32] f32
__global__ __launch_bounds__(256) void k_rnorm(const int* __restrict__ x,
                                               const float* __restrict__ emb,
                                               float* __restrict__ r, int n) {
    int t = blockIdx.x * blockDim.x + threadIdx.x;
    if (t >= n) return;
    int xv = x[t];
    const float* e = emb + (size_t)xv * 32;
    float v[32], ss = 0.f;
#pragma unroll
    for (int k = 0; k < 32; ++k) { v[k] = e[k]; ss += v[k] * v[k]; }
    float inv = fast_rcp(sqrtf(ss) + 1e-8f);
    float* o = r + (size_t)t * 32;
#pragma unroll
    for (int k = 0; k < 32; ++k) o[k] = v[k] * inv;
}

// ---------------------------------------------------------------------------
// Kernel 2: pack row-major [N,Kdim] f32 weight into WMMA-B fragment-ready f16
__global__ __launch_bounds__(256) void k_pack_b(const float* __restrict__ w,
                                                _Float16* __restrict__ outp,
                                                int Kdim, int numKT, int NTt) {
    int t = blockIdx.x * blockDim.x + threadIdx.x;
    int total = numKT * NTt * 32;
    if (t >= total) return;
    int lane = t & 31;
    int nt   = (t >> 5) % NTt;
    int kt   = (t >> 5) / NTt;
    int n    = nt * 16 + (lane & 15);
    int hi   = lane >> 4;
    FragU u;
#pragma unroll
    for (int e = 0; e < 16; ++e) {
        int col = kt * 32 + kfrag(e, hi);
        u.v[e] = (_Float16)w[(size_t)n * Kdim + col];
    }
    uint4* dst = (uint4*)(outp + (size_t)t * 16);
    dst[0] = u.q[0];
    dst[1] = u.q[1];
}

// ---------------------------------------------------------------------------
// Kernel 3: sign[i][j] = gp_table[i][j][i^j]  (the only nonzero per (i,j))
__global__ __launch_bounds__(256) void k_sign(const float* __restrict__ gp,
                                              float* __restrict__ s) {
    int t = blockIdx.x * blockDim.x + threadIdx.x;
    if (t < 1024) {
        int i = t >> 5, j = t & 31;
        s[t] = gp[((size_t)i * 32 + j) * 32 + (i ^ j)];
    }
}

// ---------------------------------------------------------------------------
// Kernel 4: persistent recurrent kernel. 8 WGs x 256 threads (8 wave32).
// 3 barriers per timestep: [gate]->B1->[sigmoid+GP+update]->B2->
// [normalize || wave0: tensorwait+stage r(t+1)+issue TDM(t+2)]->B3
__global__ __launch_bounds__(256) void k_versor_rnn(
    const float*    __restrict__ r_ws,   // [B][T][32]
    const _Float16* __restrict__ gwPack, // [65][4][32][16]
    const _Float16* __restrict__ w1Pack, // [64][8][32][16]
    const float*    __restrict__ signT,  // [32][32]
    const float*    __restrict__ gate_b, // [64]
    const float*    __restrict__ b1,     // [128]
    const float*    __restrict__ ln_g,   // [128]
    const float*    __restrict__ ln_b,   // [128]
    const float*    __restrict__ w2,     // [2][128]
    const float*    __restrict__ b2,     // [2]
    float*          __restrict__ out,    // [B][2]
    int T) {
    extern __shared__ char smem[];
    _Float16* hH  = (_Float16*)(smem + HH_OFF);  // [BB][65][32] f16 (row 64 = r_t)
    float*    hF  = (float*)(smem + HF_OFF);     // [BB][64][32] f32 master
    float*    rS  = (float*)(smem + RS_OFF);     // [2][BB][32] f32 (dbl buffer)
    float*    sS  = (float*)(smem + SS_OFF);     // [32][32] sign table
    float*    gS  = (float*)(smem + GS_OFF);     // [BB][64] gate values
    float*    gp  = (float*)(smem + GP_OFF);     // [2][BB][64] partials / z[BB][128]
    float*    gbS = (float*)(smem + GB_OFF);     // [64] gate bias mirror

    const int tid  = threadIdx.x;
    const int lane = tid & 31;
    const int wv   = tid >> 5;       // wave 0..7
    const int hi   = lane >> 4;      // K-half select
    const int ln16 = lane & 15;
    const size_t rbase = (size_t)blockIdx.x * BB * T * 32;

#if HAVE_TDM
    if (wv == 0) tdm_load_r(r_ws + rbase, RS_OFF, T * 32);  // t=0 -> buf 0 (overlaps init)
#endif
    // ---- init h0 (blade 0 = 1; hH row 64 left for r staging) + LDS mirrors ----
    for (int idx = tid; idx < BB * 64 * 32; idx += 256) {
        int b = idx >> 11, rem = idx & 2047, d = rem >> 5, k = rem & 31;
        float v = (k == 0) ? 1.0f : 0.0f;
        hF[idx] = v;
        hH[((size_t)b * 65 + d) * 32 + k] = (_Float16)v;
    }
    for (int idx = tid; idx < 1024; idx += 256) sS[idx] = signT[idx];
    if (tid < 64) gbS[tid] = gate_b[tid];

    if (wv == 0) {  // stage r(0) into hH row 64 (+ rS buf0); prefetch r(1)
#if HAVE_TDM
        __builtin_amdgcn_s_wait_tensorcnt(0);
#pragma unroll
        for (int q = 0; q < 16; ++q) {
            int idx = lane + q * 32, b = idx >> 5, k = idx & 31;
            hH[((size_t)b * 65 + 64) * 32 + k] = (_Float16)rS[idx];
        }
        if (T > 1) tdm_load_r(r_ws + rbase + 32, RS_OFF + BB * 32 * 4, T * 32);
#else
#pragma unroll
        for (int q = 0; q < 16; ++q) {
            int idx = lane + q * 32, b = idx >> 5, k = idx & 31;
            float v = r_ws[rbase + (size_t)b * T * 32 + k];
            rS[idx] = v;
            hH[((size_t)b * 65 + 64) * 32 + k] = (_Float16)v;
        }
#endif
    }
    __syncthreads();

    for (int t = 0; t < T; ++t) {
        float* rcur = rS + (t & 1) * (BB * 32);

        // ---- gate GEMM: [16,2080]@[2080,64]; two independent acc chains ----
        {
            const int nt = wv & 3;   // N tile (d block)
            const int kh = wv >> 2;  // K half
            v8f acc0 = {}, acc1 = {};
            if (kh == 0) {           // peel kt=0 so both halves loop 32x
                FragU ua = loadArow(hH, ln16 * 65 + 0, hi);
                FragU ub = loadB(gwPack, 0 * 4 + nt, lane);
                acc0 = wmma_f16(ua.v, ub.v, acc0);
            }
            const int ktb = kh ? 33 : 1;
#pragma unroll 4
            for (int s = 0; s < 32; s += 2) {
                FragU ua0 = loadArow(hH, ln16 * 65 + ktb + s, hi);
                FragU ub0 = loadB(gwPack, (ktb + s) * 4 + nt, lane);
                acc0 = wmma_f16(ua0.v, ub0.v, acc0);
                FragU ua1 = loadArow(hH, ln16 * 65 + ktb + s + 1, hi);
                FragU ub1 = loadB(gwPack, (ktb + s + 1) * 4 + nt, lane);
                acc1 = wmma_f16(ua1.v, ub1.v, acc1);
            }
#pragma unroll
            for (int r = 0; r < 8; ++r) {
                int m = r + 8 * hi;  // batch row
                gp[((size_t)kh * BB + m) * 64 + nt * 16 + ln16] = acc0[r] + acc1[r];
            }
        }
        __syncthreads();  // B1: partials complete, h mirror reads done

        // ---- sigmoid for this wave's 2 batches (same-wave producer/consumer) ----
#pragma unroll
        for (int q = 0; q < 4; ++q) {
            int idx = lane + q * 32;             // 0..127
            int b = wv * 2 + (idx >> 6), d = idx & 63;
            float s = gp[(size_t)b * 64 + d] + gp[(size_t)(BB + b) * 64 + d] + gbS[d];
            gS[b * 64 + d] = fast_rcp(1.f + __expf(-s));
        }

        // ---- geometric product GEMM + gated update (2 batches per wave) ----
        for (int bb = 0; bb < 2; ++bb) {
            int b = wv * 2 + bb;
            // B fragments of M^T in registers: B[j][k] = sign[k^j][j]*r[k^j]
            FragU bf0, bf1;
#pragma unroll
            for (int e = 0; e < 16; ++e) {
                int j  = kfrag(e, hi);
                int i0 = ln16 ^ j, i1 = (16 + ln16) ^ j;
                bf0.v[e] = (_Float16)(rcur[b * 32 + i0] * sS[i0 * 32 + j]);
                bf1.v[e] = (_Float16)(rcur[b * 32 + i1] * sS[i1 * 32 + j]);
            }
#pragma unroll
            for (int mt = 0; mt < 4; ++mt) {
                FragU ua = loadArow(hH, b * 65 + mt * 16 + ln16, hi);
                v8f c0 = {}, c1 = {};
                c0 = wmma_f16(ua.v, bf0.v, c0);
                c1 = wmma_f16(ua.v, bf1.v, c1);
#pragma unroll
                for (int r = 0; r < 8; ++r) {
                    int d = mt * 16 + r + 8 * hi;
                    float gv = gS[b * 64 + d];
                    float* hr = &hF[((size_t)b * 64 + d) * 32];
                    hr[ln16]      = (1.f - gv) * hr[ln16]      + gv * c0[r];
                    hr[16 + ln16] = (1.f - gv) * hr[16 + ln16] + gv * c1[r];
                }
            }
        }
        __syncthreads();  // B2: hF updated everywhere; rcur reads done

        // ---- normalize (all waves)  ||  wave0 also stages r(t+1) ----
        for (int row = tid; row < BB * 64; row += 256) {
            float* hr = &hF[(size_t)row * 32];
            float ss = 0.f;
#pragma unroll
            for (int k = 0; k < 32; ++k) ss += hr[k] * hr[k];
            float inv = fast_rcp(sqrtf(ss) + 1e-8f);
            int b = row >> 6, d = row & 63;
            _Float16* hh = &hH[((size_t)b * 65 + d) * 32];
#pragma unroll
            for (int k = 0; k < 32; ++k) {
                float v = hr[k] * inv;
                hr[k] = v;
                hh[k] = (_Float16)v;
            }
        }
        if (wv == 0 && (t + 1) < T) {
            float* rnext = rS + ((t + 1) & 1) * (BB * 32);
#if HAVE_TDM
            __builtin_amdgcn_s_wait_tensorcnt(0);   // TDM(t+1) complete
#pragma unroll
            for (int q = 0; q < 16; ++q) {
                int idx = lane + q * 32, b = idx >> 5, k = idx & 31;
                hH[((size_t)b * 65 + 64) * 32 + k] = (_Float16)rnext[idx];
            }
            if ((t + 2) < T)   // buf (t&1): its readers finished at B2
                tdm_load_r(r_ws + rbase + (size_t)(t + 2) * 32,
                           RS_OFF + (((t + 2) & 1) ? BB * 32 * 4 : 0), T * 32);
#else
#pragma unroll
            for (int q = 0; q < 16; ++q) {
                int idx = lane + q * 32, b = idx >> 5, k = idx & 31;
                float v = r_ws[rbase + ((size_t)b * T + (t + 1)) * 32 + k];
                rnext[idx] = v;
                hH[((size_t)b * 65 + 64) * 32 + k] = (_Float16)v;
            }
#endif
        }
        if (t == T - 8) {  // warm L2/L0 for the epilogue weights
#pragma unroll
            for (int q = 0; q < 16; ++q)
                __builtin_prefetch(w1Pack + ((size_t)tid * 16 + q) * 64, 0, 0);
        }
        __syncthreads();  // B3: h mirror + r(t+1) staged
    }

    // ---- epilogue: z = h_flat @ w1^T (M=16,N=128,K=2048; wave w owns nt=w) ----
    {
        const int nt = wv;
        v8f acc0 = {}, acc1 = {};
#pragma unroll 4
        for (int kt = 0; kt < 64; kt += 2) {
            FragU ua0 = loadArow(hH, ln16 * 65 + kt, hi);
            FragU ub0 = loadB(w1Pack, kt * 8 + nt, lane);
            acc0 = wmma_f16(ua0.v, ub0.v, acc0);
            FragU ua1 = loadArow(hH, ln16 * 65 + kt + 1, hi);
            FragU ub1 = loadB(w1Pack, (kt + 1) * 8 + nt, lane);
            acc1 = wmma_f16(ua1.v, ub1.v, acc1);
        }
        float* zS = gp;  // reuse as [BB][128]
#pragma unroll
        for (int r = 0; r < 8; ++r) {
            int m = r + 8 * hi;
            zS[(size_t)m * 128 + nt * 16 + ln16] = acc0[r] + acc1[r];
        }
    }
    __syncthreads();

    // ---- LayerNorm(128) + ReLU + w2: one thread per batch row ----
    if (tid < BB) {
        float* zr = &gp[(size_t)tid * 128];
        float mu = 0.f, m2 = 0.f;
        for (int i = 0; i < 128; ++i) {
            float z = zr[i] + b1[i];
            mu += z; m2 += z * z;
        }
        mu *= (1.f / 128.f);
        float var = m2 * (1.f / 128.f) - mu * mu;
        float inv = rsqrtf(var + 1e-5f);
        float o0 = 0.f, o1 = 0.f;
        for (int i = 0; i < 128; ++i) {
            float z = (zr[i] + b1[i] - mu) * inv * ln_g[i] + ln_b[i];
            z = fmaxf(z, 0.f);
            o0 += z * w2[i];
            o1 += z * w2[128 + i];
        }
        int gb = blockIdx.x * BB + tid;
        out[gb * 2 + 0] = o0 + b2[0];
        out[gb * 2 + 1] = o1 + b2[1];
    }
}

// ---------------------------------------------------------------------------
extern "C" void kernel_launch(void* const* d_in, const int* in_sizes, int n_in,
                              void* d_out, int out_size, void* d_ws, size_t ws_size,
                              hipStream_t stream) {
    (void)in_sizes; (void)n_in; (void)out_size; (void)ws_size;
    const int*   x       = (const int*)d_in[0];
    const float* emb     = (const float*)d_in[1];
    const float* gate_w  = (const float*)d_in[2];
    const float* gate_b  = (const float*)d_in[3];
    const float* w1      = (const float*)d_in[4];
    const float* b1      = (const float*)d_in[5];
    const float* ln_g    = (const float*)d_in[6];
    const float* ln_b    = (const float*)d_in[7];
    const float* w2      = (const float*)d_in[8];
    const float* b2      = (const float*)d_in[9];
    const float* gp_tab  = (const float*)d_in[10];
    float* out = (float*)d_out;

    // workspace layout
    char* ws = (char*)d_ws;
    float*    r_ws   = (float*)ws;                               // 8 MB
    size_t    off    = (size_t)NB * NT_SEQ * 32 * sizeof(float);
    _Float16* gwPack = (_Float16*)(ws + off);                    // 65*4*32*16 f16
    off += (size_t)65 * 4 * 32 * 16 * sizeof(_Float16);
    _Float16* w1Pack = (_Float16*)(ws + off);                    // 64*8*32*16 f16
    off += (size_t)64 * 8 * 32 * 16 * sizeof(_Float16);
    float*    signT  = (float*)(ws + off);                       // 32*32 f32

    k_rnorm<<<(NB * NT_SEQ + 255) / 256, 256, 0, stream>>>(x, emb, r_ws, NB * NT_SEQ);
    k_pack_b<<<(65 * 4 * 32 + 255) / 256, 256, 0, stream>>>(gate_w, gwPack, 2080, 65, 4);
    k_pack_b<<<(64 * 8 * 32 + 255) / 256, 256, 0, stream>>>(w1, w1Pack, 2048, 64, 8);
    k_sign<<<4, 256, 0, stream>>>(gp_tab, signT);

    hipFuncSetAttribute((const void*)k_versor_rnn,
                        hipFuncAttributeMaxDynamicSharedMemorySize, (int)SMEM_BYTES);
    k_versor_rnn<<<NWG, 256, SMEM_BYTES, stream>>>(
        r_ws, gwPack, w1Pack, signT, gate_b, b1, ln_g, ln_b, w2, b2, out, NT_SEQ);
}
